// Gemma3nAudioAttention_64596308132296
// MI455X (gfx1250) — matrix operand order repaired and checked by reading, not compiled, not run on hardware
//
#include <hip/hip_runtime.h>

#define AS1 __attribute__((address_space(1)))
#define AS3 __attribute__((address_space(3)))

typedef __bf16        v16bf __attribute__((ext_vector_type(16)));
typedef float         v8f   __attribute__((ext_vector_type(8)));
typedef unsigned int  u32x4 __attribute__((ext_vector_type(4)));
typedef int           v4i_g __attribute__((vector_size(16)));   // gcc-style, matches builtin

#if __has_builtin(__builtin_amdgcn_global_load_async_to_lds_b128) && \
    __has_builtin(__builtin_amdgcn_s_wait_asynccnt)
#define HAVE_ASYNC_LDS 1
#else
#define HAVE_ASYNC_LDS 0
#endif

// ---- problem constants ----
constexpr int B_    = 8;
constexpr int T_    = 2040;
constexpr int H_    = 1536;
constexpr int NH_   = 8;
constexpr int HD_   = 192;
constexpr int CHUNK_= 12;
constexpr int PAST_ = 12;
constexpr int CTXS_ = 24;    // CHUNK + PAST
constexpr int FSPAN_= 13;    // PAST + 1
constexpr int U_    = T_ / CHUNK_;   // 170
constexpr int M_    = B_ * T_;       // 16320
constexpr int NF_   = 3 * H_;        // 4608 fused q|k|v
constexpr float CAP_ = 50.0f;

__device__ inline int imin(int a, int b) { return a < b ? a : b; }
__device__ inline int imax(int a, int b) { return a > b ? a : b; }
__device__ inline v8f vzero8() {
  v8f r;
  for (int i = 0; i < 8; ++i) r[i] = 0.0f;
  return r;
}

union Frag16 { v16bf v; u32x4 q[2]; };

__device__ inline void copy16B_to_lds(const __bf16* g, __bf16* l) {
#if HAVE_ASYNC_LDS
  __builtin_amdgcn_global_load_async_to_lds_b128((AS1 v4i_g*)g, (AS3 v4i_g*)l,
                                                 0, 0);
#else
  *(u32x4*)l = *(const u32x4*)g;
#endif
}

// ---------------------------------------------------------------------------
// Pack kernels: f32 -> bf16 (weights pre-transposed [N][K], q-scale folded in)
// ---------------------------------------------------------------------------
__global__ void pack_x_kernel(const float* __restrict__ x,
                              __bf16* __restrict__ xb, int n) {
  int i = blockIdx.x * blockDim.x + threadIdx.x;
  int stride = gridDim.x * blockDim.x;
  for (; i < n; i += stride) xb[i] = (__bf16)x[i];
}

__global__ void pack_w_kernel(const float* __restrict__ Wq,
                              const float* __restrict__ Wk,
                              const float* __restrict__ Wv,
                              const float* __restrict__ pds,
                              __bf16* __restrict__ wt) {
  // wt layout: [NF_][H_]  (N-major, K contiguous)
  int i = blockIdx.x * blockDim.x + threadIdx.x;
  int total = NF_ * H_;
  if (i >= total) return;
  int nn = i / H_;       // fused output column 0..4607
  int k  = i % H_;       // input feature
  int which = nn / H_;   // 0=q 1=k 2=v
  int j = nn % H_;       // column within q/k/v
  float v;
  if (which == 0) {
    float p  = pds[j % HD_];
    float sp = log1pf(expf(p));                       // softplus
    float scale = rsqrtf((float)HD_) * 1.44269504088896340736f * sp; // *1/ln2
    v = Wq[k * H_ + j] * scale;
  } else if (which == 1) {
    v = Wk[k * H_ + j];
  } else {
    v = Wv[k * H_ + j];
  }
  wt[i] = (__bf16)v;
}

// ---------------------------------------------------------------------------
// Relative sinusoidal position embedding: sin_emb (13,1536), then @ Wpos
// ---------------------------------------------------------------------------
__global__ void sinemb_kernel(float* __restrict__ se) {
  int i = blockIdx.x * blockDim.x + threadIdx.x;
  if (i >= FSPAN_ * H_) return;
  int f = i / H_, k = i % H_;
  float pos = (float)(PAST_ - f);              // 12 .. 0
  int kk = (k < H_ / 2) ? k : k - H_ / 2;
  float invts = expf(-(float)kk * (logf(10000.0f) / (float)(H_ / 2 - 1)));
  float st = pos * invts;
  se[i] = (k < H_ / 2) ? sinf(st) : cosf(st);
}

__global__ void posproj_kernel(const float* __restrict__ se,
                               const float* __restrict__ Wpos,
                               float* __restrict__ pe) {
  int i = blockIdx.x * blockDim.x + threadIdx.x;
  if (i >= FSPAN_ * H_) return;
  int f = i / H_, j = i % H_;
  float acc = 0.0f;
  for (int k = 0; k < H_; ++k) acc += se[f * H_ + k] * Wpos[k * H_ + j];
  pe[i] = acc;   // pe[f][n*HD + h]
}

// ---------------------------------------------------------------------------
// Fused QKV GEMM: C[M][4608] = Xb[M][1536] * Wt^T, bf16 in / f32 acc / bf16 out
// 64x128 tile per 128-thread block (4 waves, each 32x64 = 2x4 WMMA tiles)
// ---------------------------------------------------------------------------
#define GBM 64
#define GBN 128
#define GBK 32

__global__ __launch_bounds__(128) void qkv_gemm_kernel(
    const __bf16* __restrict__ X, const __bf16* __restrict__ Wt,
    __bf16* __restrict__ C) {
  __shared__ __align__(16) __bf16 lA[GBM * GBK];
  __shared__ __align__(16) __bf16 lB[GBN * GBK];

  const int m0 = blockIdx.x * GBM;
  const int n0 = blockIdx.y * GBN;
  const int tid = threadIdx.x;
  const int lane = tid & 31;
  const int wave = tid >> 5;
  const int wm = (wave & 1) * 32;    // wave M offset within block tile
  const int wn = (wave >> 1) * 64;   // wave N offset within block tile
  const int l15 = lane & 15;
  const int lhalf = lane >> 4;

  v8f acc[2][4];
  for (int mt = 0; mt < 2; ++mt)
    for (int nt = 0; nt < 4; ++nt) acc[mt][nt] = vzero8();

  for (int k0 = 0; k0 < H_; k0 += GBK) {
    // --- stage A tile (64x32) : 256 x 16B chunks, 2 per thread ---
#pragma unroll
    for (int r = 0; r < 2; ++r) {
      int i = tid + 128 * r;
      int row = i >> 2, seg = i & 3;
      copy16B_to_lds(X + (size_t)(m0 + row) * H_ + k0 + seg * 8,
                     lA + row * GBK + seg * 8);
    }
    // --- stage B tile (128x32, N-major) : 512 x 16B chunks, 4 per thread ---
#pragma unroll
    for (int r = 0; r < 4; ++r) {
      int i = tid + 128 * r;
      int row = i >> 2, seg = i & 3;
      copy16B_to_lds(Wt + (size_t)(n0 + row) * H_ + k0 + seg * 8,
                     lB + row * GBK + seg * 8);
    }
    if (k0 + GBK < H_) {   // hint next k-slice toward the caches
      __builtin_prefetch(X + (size_t)(m0 + (tid >> 1)) * H_ + k0 + GBK, 0, 1);
      __builtin_prefetch(Wt + (size_t)(n0 + tid) * H_ + k0 + GBK, 0, 1);
    }
#if HAVE_ASYNC_LDS
    __builtin_amdgcn_s_wait_asynccnt(0);
#endif
    __syncthreads();

    // --- per-lane fragments (ISA 7.12.2 layouts) ---
    Frag16 a[2], b[4];
#pragma unroll
    for (int mt = 0; mt < 2; ++mt) {
      int row = wm + mt * 16 + l15;
      int kb0 = lhalf * 8;
      a[mt].q[0] = *(const u32x4*)(lA + row * GBK + kb0);        // K = kb0..kb0+7
      a[mt].q[1] = *(const u32x4*)(lA + row * GBK + 16 + kb0);   // K = 16+kb0..
    }
#pragma unroll
    for (int nt = 0; nt < 4; ++nt) {
      int col = wn + nt * 16 + l15;
      int kb = lhalf * 8;
      b[nt].q[0] = *(const u32x4*)(lB + col * GBK + kb);
      b[nt].q[1] = *(const u32x4*)(lB + col * GBK + kb + 16);
    }
#pragma unroll
    for (int mt = 0; mt < 2; ++mt)
#pragma unroll
      for (int nt = 0; nt < 4; ++nt)
        acc[mt][nt] = __builtin_amdgcn_wmma_f32_16x16x32_bf16(
            false, a[mt].v, false, b[nt].v, (short)0, acc[mt][nt], false, false);
    __syncthreads();
  }

  // --- epilogue: C layout lane -> (row = r + 8*lhalf, col = l15) ---
#pragma unroll
  for (int mt = 0; mt < 2; ++mt) {
#pragma unroll
    for (int nt = 0; nt < 4; ++nt) {
      int colg = n0 + wn + nt * 16 + l15;
#pragma unroll
      for (int r = 0; r < 8; ++r) {
        int rowg = m0 + wm + mt * 16 + lhalf * 8 + r;
        C[(size_t)rowg * NF_ + colg] = (__bf16)acc[mt][nt][r];
      }
    }
  }
}

// ---------------------------------------------------------------------------
// Local chunked attention: one wave per (b, chunk u, head n)
// scores = q*k^T via WMMA (12->16 rows, 24 cols = 2 tiles, K=192 = 6 steps)
// ---------------------------------------------------------------------------
__global__ __launch_bounds__(32) void attn_kernel(
    const __bf16* __restrict__ QKV, const float* __restrict__ pe,
    const unsigned char* __restrict__ mask, float* __restrict__ out) {
  const int u = blockIdx.x, hn = blockIdx.y, b = blockIdx.z;
  const int lane = threadIdx.x;
  const int l15 = lane & 15;
  const int lhalf = lane >> 4;

  __shared__ float s_bd[CHUNK_ * FSPAN_];
  __shared__ float s_lg[CHUNK_][CTXS_];
  __shared__ float s_p[CHUNK_][CTXS_];

  const size_t base = (size_t)b * T_ * NF_;
  const int qoff = hn * HD_;
  const int koff = H_ + hn * HD_;
  const int voff = 2 * H_ + hn * HD_;

  // ---- term_ac via WMMA ----
  v8f acc0 = vzero8(), acc1 = vzero8();
  const int wq = imin(l15, CHUNK_ - 1);  // pad rows 12..15 with row 11 (unused)
  const __bf16* qrow = QKV + base + (size_t)(u * CHUNK_ + wq) * NF_ + qoff;
  const int c0 = l15, c1 = 16 + l15;
  const int tk0 = imin(imax(u * CHUNK_ + c0 - PAST_, 0), T_ - 1);
  const int tk1 = imin(imax(u * CHUNK_ + c1 - PAST_, 0), T_ - 1);
  const __bf16* krow0 = QKV + base + (size_t)tk0 * NF_ + koff;
  const __bf16* krow1 = QKV + base + (size_t)tk1 * NF_ + koff;

#pragma unroll
  for (int ks = 0; ks < HD_ / 32; ++ks) {
    Frag16 a, f0, f1;
    int kb0 = ks * 32 + lhalf * 8;
    a.q[0] = *(const u32x4*)(qrow + kb0);
    a.q[1] = *(const u32x4*)(qrow + kb0 + 16);
    int kb = ks * 32 + lhalf * 8;
    f0.q[0] = *(const u32x4*)(krow0 + kb);
    f0.q[1] = *(const u32x4*)(krow0 + kb + 16);
    f1.q[0] = *(const u32x4*)(krow1 + kb);
    f1.q[1] = *(const u32x4*)(krow1 + kb + 16);
    acc0 = __builtin_amdgcn_wmma_f32_16x16x32_bf16(false, a.v, false, f0.v,
                                                   (short)0, acc0, false, false);
    acc1 = __builtin_amdgcn_wmma_f32_16x16x32_bf16(false, a.v, false, f1.v,
                                                   (short)0, acc1, false, false);
  }

  // ---- term_bd: bd[w][f] = q[w] . pe[f]  (156 dot products over 192) ----
  for (int idx = lane; idx < CHUNK_ * FSPAN_; idx += 32) {
    int w = idx / FSPAN_, f = idx % FSPAN_;
    const __bf16* qr = QKV + base + (size_t)(u * CHUNK_ + w) * NF_ + qoff;
    const float* pr = pe + f * H_ + hn * HD_;
    float s = 0.0f;
    for (int h = 0; h < HD_; ++h) s += (float)qr[h] * pr[h];
    s_bd[idx] = s;
  }

  // ---- spill WMMA C tiles to LDS: row = r + 8*lhalf, col = l15 ----
#pragma unroll
  for (int r = 0; r < 8; ++r) {
    int w = r + lhalf * 8;
    if (w < CHUNK_) {
      s_lg[w][l15] = acc0[r];
      if (16 + l15 < CTXS_) s_lg[w][16 + l15] = acc1[r];
    }
  }
  __syncthreads();

  // ---- mask + rel-shift + tanh cap + softmax (one lane per query row) ----
  if (lane < CHUNK_) {
    const int w = lane;
    float lg[CTXS_];
    float mx = -3.4e38f;
#pragma unroll
    for (int c = 0; c < CTXS_; ++c) {
      int tk = u * CHUNK_ + c - PAST_;
      bool valid = (tk >= 0) && (tk < T_) && (mask[b * T_ + tk] == 0);
      bool causal = (c >= w) && (c <= w + PAST_);
      float l = s_lg[w][c];
      int d = c - w;
      if (d >= 0 && d < FSPAN_) l += s_bd[w * FSPAN_ + d];  // relative shift
      l = tanhf(l * (1.0f / CAP_)) * CAP_;
      l = (valid && causal) ? l : -1e30f;
      lg[c] = l;
      mx = fmaxf(mx, l);
    }
    float sum = 0.0f;
#pragma unroll
    for (int c = 0; c < CTXS_; ++c) {
      float e = expf(lg[c] - mx);
      lg[c] = e;
      sum += e;
    }
    float inv = 1.0f / sum;
#pragma unroll
    for (int c = 0; c < CTXS_; ++c) s_p[w][c] = lg[c] * inv;
  }
  __syncthreads();

  // ---- PV: each lane owns 6 head dims for all 12 rows ----
  const int h0 = lane * (HD_ / 32);  // 6 dims per lane
  float po[CHUNK_][HD_ / 32];
#pragma unroll
  for (int w = 0; w < CHUNK_; ++w)
#pragma unroll
    for (int j = 0; j < HD_ / 32; ++j) po[w][j] = 0.0f;

  for (int c = 0; c < CTXS_; ++c) {
    int tk = imin(imax(u * CHUNK_ + c - PAST_, 0), T_ - 1);
    const __bf16* vr = QKV + base + (size_t)tk * NF_ + voff + h0;
    float vv[HD_ / 32];
#pragma unroll
    for (int j = 0; j < HD_ / 32; ++j) vv[j] = (float)vr[j];
#pragma unroll
    for (int w = 0; w < CHUNK_; ++w) {
      float p = s_p[w][c];
#pragma unroll
      for (int j = 0; j < HD_ / 32; ++j) po[w][j] += p * vv[j];
    }
  }
#pragma unroll
  for (int w = 0; w < CHUNK_; ++w) {
    float* op = out + ((size_t)(b * T_ + u * CHUNK_ + w) * NH_ + hn) * HD_ + h0;
#pragma unroll
    for (int j = 0; j < HD_ / 32; ++j) op[j] = po[w][j];
  }
}

// ---------------------------------------------------------------------------
extern "C" void kernel_launch(void* const* d_in, const int* in_sizes, int n_in,
                              void* d_out, int out_size, void* d_ws,
                              size_t ws_size, hipStream_t stream) {
  const float* hs   = (const float*)d_in[0];
  const float* Wq   = (const float*)d_in[1];
  const float* Wk   = (const float*)d_in[2];
  const float* Wv   = (const float*)d_in[3];
  const float* Wpos = (const float*)d_in[4];
  const float* pds  = (const float*)d_in[5];
  const unsigned char* mask = (const unsigned char*)d_in[6];
  float* out = (float*)d_out;

  char* ws = (char*)d_ws;
  size_t off = 0;
  auto take = [&](size_t bytes) {
    size_t o = off;
    off = (off + bytes + 255) & ~(size_t)255;
    return o;
  };
  __bf16* Xb  = (__bf16*)(ws + take((size_t)M_ * H_ * sizeof(__bf16)));
  __bf16* Wt  = (__bf16*)(ws + take((size_t)NF_ * H_ * sizeof(__bf16)));
  __bf16* QKV = (__bf16*)(ws + take((size_t)M_ * NF_ * sizeof(__bf16)));
  float*  se  = (float*)(ws + take((size_t)FSPAN_ * H_ * sizeof(float)));
  float*  pe  = (float*)(ws + take((size_t)FSPAN_ * H_ * sizeof(float)));
  (void)ws_size; (void)in_sizes; (void)n_in; (void)out_size;

  pack_x_kernel<<<2048, 256, 0, stream>>>(hs, Xb, M_ * H_);
  pack_w_kernel<<<(NF_ * H_ + 255) / 256, 256, 0, stream>>>(Wq, Wk, Wv, pds, Wt);
  sinemb_kernel<<<(FSPAN_ * H_ + 255) / 256, 256, 0, stream>>>(se);
  posproj_kernel<<<(FSPAN_ * H_ + 255) / 256, 256, 0, stream>>>(se, Wpos, pe);
  qkv_gemm_kernel<<<dim3(M_ / GBM, NF_ / GBN), 128, 0, stream>>>(Xb, Wt, QKV);
  attn_kernel<<<dim3(U_, NH_, B_), 32, 0, stream>>>(QKV, pe, mask, out);
}
